// Mamba3DBlock_3006477107886
// MI455X (gfx1250) — matrix operand order repaired and checked by reading, not compiled, not run on hardware
//
#include <hip/hip_runtime.h>
#include <hip/hip_bf16.h>
#include <math.h>

typedef __attribute__((ext_vector_type(16))) __bf16 v16bf;
typedef __attribute__((ext_vector_type(8)))  __bf16 v8bf;
typedef __attribute__((ext_vector_type(8)))  float  v8f;

#define C_DIM 384
#define NPTS 2048
#define BATCH 4
#define ROWS_ALL (BATCH * (NPTS + 1))   // 8196
#define ROWS_NO  (BATCH * NPTS)         // 8192
#define KG 8

union V16U { v16bf v; v8bf h[2]; };

__device__ __forceinline__ float gelu_exact(float v) {
    return 0.5f * v * (1.0f + erff(v * 0.7071067811865475f));
}

__device__ __forceinline__ v16bf load_frag16(const __bf16* p0, const __bf16* p1) {
    V16U u;
    u.h[0] = *(const v8bf*)p0;
    u.h[1] = *(const v8bf*)p1;
    return u.v;
}

// ---------------- LayerNorm (one block per row, C=384, 128 threads) ----------------
template<bool BF16OUT>
__global__ void ln_kernel(const float* __restrict__ x, const float* __restrict__ g,
                          const float* __restrict__ b, float* __restrict__ outF,
                          __bf16* __restrict__ outB) {
    __shared__ float s1[128], s2[128];
    __shared__ float mv[2];
    const int row = blockIdx.x;
    const int tid = threadIdx.x;
    const float* xr = x + (size_t)row * C_DIM;
    float v0 = xr[tid], v1 = xr[tid + 128], v2 = xr[tid + 256];
    s1[tid] = v0 + v1 + v2;
    s2[tid] = v0 * v0 + v1 * v1 + v2 * v2;
    __syncthreads();
    for (int o = 64; o > 0; o >>= 1) {
        if (tid < o) { s1[tid] += s1[tid + o]; s2[tid] += s2[tid + o]; }
        __syncthreads();
    }
    if (tid == 0) {
        float m = s1[0] * (1.0f / C_DIM);
        float var = s2[0] * (1.0f / C_DIM) - m * m;
        mv[0] = m;
        mv[1] = rsqrtf(var + 1e-5f);
    }
    __syncthreads();
    float m = mv[0], r = mv[1];
    #pragma unroll
    for (int i = tid; i < C_DIM; i += 128) {
        float v = (xr[i] - m) * r * g[i] + b[i];
        if (BF16OUT) outB[(size_t)row * C_DIM + i] = (__bf16)v;
        else         outF[(size_t)row * C_DIM + i] = v;
    }
}

// ---------------- kNN top-8 (block = 256 queries, centers in LDS) ----------------
__global__ void knn_kernel(const float* __restrict__ center, int* __restrict__ idx) {
    __shared__ float cx[NPTS], cy[NPTS], cz[NPTS], sq[NPTS];
    const int b = blockIdx.y, tid = threadIdx.x;
    const float* cb = center + (size_t)b * NPTS * 3;
    for (int i = tid; i < NPTS; i += 256) {
        float x = cb[i * 3 + 0], y = cb[i * 3 + 1], z = cb[i * 3 + 2];
        cx[i] = x; cy[i] = y; cz[i] = z;
        sq[i] = x * x + y * y + z * z;
    }
    __syncthreads();
    const int n = blockIdx.x * 256 + tid;
    const float qx = cx[n], qy = cy[n], qz = cz[n], qs = sq[n];
    float bd[KG];
    int   bi[KG];
    #pragma unroll
    for (int j = 0; j < KG; j++) { bd[j] = 3.4e38f; bi[j] = 0; }
    for (int m = 0; m < NPTS; m++) {
        float d = qs + sq[m] - 2.0f * (qx * cx[m] + qy * cy[m] + qz * cz[m]);
        if (d < bd[KG - 1]) {
            int j = KG - 1;
            while (j > 0 && bd[j - 1] > d) {       // strict > keeps stable tie order
                bd[j] = bd[j - 1]; bi[j] = bi[j - 1]; j--;
            }
            bd[j] = d; bi[j] = m;
        }
    }
    int* op = idx + (size_t)(b * NPTS + n) * KG;
    #pragma unroll
    for (int j = 0; j < KG; j++) op[j] = bi[j];
}

// ---------------- knn gather: S = sum_k diff, plus partial (sum, sumsq) ----------------
__global__ void gather_kernel(const float* __restrict__ xn, const int* __restrict__ idx,
                              float* __restrict__ S, float* __restrict__ part) {
    __shared__ float s1[512], s2[512];
    const int bn = blockIdx.x;
    const int b = bn >> 11;
    const int n = bn & (NPTS - 1);
    const int c = threadIdx.x;                       // 384 threads
    const float* base = xn + (size_t)b * (NPTS + 1) * C_DIM;
    const float xno = base[(size_t)(n + 1) * C_DIM + c];
    const int* ip = idx + (size_t)bn * KG;
    float s = 0.f, ss = 0.f;
    #pragma unroll
    for (int k = 0; k < KG; k++) {
        float v = base[(size_t)(ip[k] + 1) * C_DIM + c] - xno;
        s += v; ss += v * v;
    }
    S[(size_t)bn * C_DIM + c] = s;
    s1[c] = s; s2[c] = ss;
    if (c < 128) { s1[c + 384] = 0.f; s2[c + 384] = 0.f; }
    __syncthreads();
    for (int o = 256; o > 0; o >>= 1) {
        if (c < o) { s1[c] += s1[c + o]; s2[c] += s2[c + o]; }
        __syncthreads();
    }
    if (c == 0) { part[bn] = s1[0]; part[ROWS_NO + bn] = s2[0]; }
}

// ---------------- global std (deterministic, single block) ----------------
__global__ void std_kernel(const float* __restrict__ part, float* __restrict__ scal) {
    __shared__ double d1[256], d2[256];
    const int tid = threadIdx.x;
    double s = 0.0, ss = 0.0;
    for (int i = tid; i < ROWS_NO; i += 256) { s += part[i]; ss += part[ROWS_NO + i]; }
    d1[tid] = s; d2[tid] = ss;
    __syncthreads();
    for (int o = 128; o > 0; o >>= 1) {
        if (tid < o) { d1[tid] += d1[tid + o]; d2[tid] += d2[tid + o]; }
        __syncthreads();
    }
    if (tid == 0) {
        const double M = (double)ROWS_NO * KG * C_DIM;
        double var = (d2[0] - d1[0] * d1[0] / M) / (M - 1.0);
        float stdv = (float)sqrt(var);
        scal[0] = 1.0f / (stdv + 1e-5f);
    }
}

// ---------------- ef build (bf16): [alpha1*mean_k(diff)*inv+beta1 , alpha2*xno+beta2] ----------------
__global__ void ef_kernel(const float* __restrict__ S, const float* __restrict__ xn,
                          const float* __restrict__ alpha, const float* __restrict__ beta,
                          const float* __restrict__ scal, __bf16* __restrict__ ef) {
    const int t = blockIdx.x * 256 + threadIdx.x;
    if (t >= ROWS_NO * C_DIM) return;
    const int row = t / C_DIM, c = t - row * C_DIM;
    const int b = row >> 11, n = row & (NPTS - 1);
    const float inv = scal[0];
    const float xno = xn[((size_t)(b * (NPTS + 1) + 1 + n)) * C_DIM + c];
    const float m1 = S[t] * (1.0f / KG) * inv;
    ef[(size_t)row * (2 * C_DIM) + c]         = (__bf16)(alpha[c] * m1 + beta[c]);
    ef[(size_t)row * (2 * C_DIM) + C_DIM + c] = (__bf16)(alpha[C_DIM + c] * xno + beta[C_DIM + c]);
}

// ---------------- weight convert+transpose: (Kd x Nd f32) -> (Nd x Kd bf16) ----------------
__global__ void wconv_kernel(const float* __restrict__ w, __bf16* __restrict__ wT,
                             int Kd, int Nd) {
    const int t = blockIdx.x * 256 + threadIdx.x;
    if (t >= Kd * Nd) return;
    const int k = t / Nd, n = t - k * Nd;
    wT[(size_t)n * Kd + k] = (__bf16)w[t];
}

// ---------------- cls-row residual: x2[b,0,:] = x[b,0,:] + xn[b,0,:] ----------------
__global__ void cls_kernel(const float* __restrict__ x, const float* __restrict__ xn,
                           float* __restrict__ x2) {
    const int t = blockIdx.x * 256 + threadIdx.x;
    if (t >= BATCH * C_DIM) return;
    const int b = t / C_DIM, c = t - b * C_DIM;
    const size_t o = (size_t)b * (NPTS + 1) * C_DIM + c;
    x2[o] = x[o] + xn[o];
}

// ---------------- WMMA bf16 GEMM: block=128 thr (4 waves), wave = 32 rows x 64 cols ----------------
// Block tile = 128 rows x 64 cols. A: M x K bf16 row-major; Bt: N x K bf16 (N-major); N = gridDim.x*64.
// Per K-tile/wave: 4 A b128-loads + 8 B b128-loads feed 8 WMMAs (1.5 vmem per wmma).
// MODE 1: outB = bf16 gelu(acc + bias)
// MODE 2: outF[b,1+n,c] = resid[b,1+n,c] + acc + bias   (M rows map to (b,n); resid/out are Bx2049xC)
// MODE 4: outF[row,c]   = resid[row,c]   + acc + bias
template<int MODE>
__global__ void gemm_kernel(const __bf16* __restrict__ A, const __bf16* __restrict__ Bt,
                            const float* __restrict__ bias, int M, int K,
                            __bf16* __restrict__ outB, float* __restrict__ outF,
                            const float* __restrict__ resid) {
    const int lane = threadIdx.x & 31;
    const int wave = threadIdx.x >> 5;
    const int half = lane >> 4;
    const int l16  = lane & 15;
    const int row0 = blockIdx.y * 128 + wave * 32;
    const int col0 = blockIdx.x * 64;
    const int N    = gridDim.x * 64;

    const int arow0 = row0 + l16;
    const int arow1 = row0 + 16 + l16;
    const int ar0c = (arow0 < M) ? arow0 : (M - 1);   // clamp: keeps EXEC uniform for WMMA
    const int ar1c = (arow1 < M) ? arow1 : (M - 1);
    const __bf16* ap0 = A + (size_t)ar0c * K;
    const __bf16* ap1 = A + (size_t)ar1c * K;

    const __bf16* bp0 = Bt + (size_t)(col0 +  0 + l16) * K + half * 16;
    const __bf16* bp1 = Bt + (size_t)(col0 + 16 + l16) * K + half * 16;
    const __bf16* bp2 = Bt + (size_t)(col0 + 32 + l16) * K + half * 16;
    const __bf16* bp3 = Bt + (size_t)(col0 + 48 + l16) * K + half * 16;

    v8f acc[8] = {};
    for (int kt = 0; kt < K; kt += 32) {
        // A frags (16x32 bf16 each): half0 -> K=kt..kt+15 ; half1 -> kt+16..kt+31
        v16bf a0 = load_frag16(ap0 + kt + half * 16, ap0 + kt + half * 16 + 8);
        v16bf a1 = load_frag16(ap1 + kt + half * 16, ap1 + kt + half * 16 + 8);
        v16bf b0 = load_frag16(bp0 + kt, bp0 + kt + 8);
        v16bf b1 = load_frag16(bp1 + kt, bp1 + kt + 8);
        v16bf b2 = load_frag16(bp2 + kt, bp2 + kt + 8);
        v16bf b3 = load_frag16(bp3 + kt, bp3 + kt + 8);
        acc[0] = __builtin_amdgcn_wmma_f32_16x16x32_bf16(false, a0, false, b0, (short)0, acc[0], false, false);
        acc[4] = __builtin_amdgcn_wmma_f32_16x16x32_bf16(false, a1, false, b0, (short)0, acc[4], false, false);
        acc[1] = __builtin_amdgcn_wmma_f32_16x16x32_bf16(false, a0, false, b1, (short)0, acc[1], false, false);
        acc[5] = __builtin_amdgcn_wmma_f32_16x16x32_bf16(false, a1, false, b1, (short)0, acc[5], false, false);
        acc[2] = __builtin_amdgcn_wmma_f32_16x16x32_bf16(false, a0, false, b2, (short)0, acc[2], false, false);
        acc[6] = __builtin_amdgcn_wmma_f32_16x16x32_bf16(false, a1, false, b2, (short)0, acc[6], false, false);
        acc[3] = __builtin_amdgcn_wmma_f32_16x16x32_bf16(false, a0, false, b3, (short)0, acc[3], false, false);
        acc[7] = __builtin_amdgcn_wmma_f32_16x16x32_bf16(false, a1, false, b3, (short)0, acc[7], false, false);
    }

    #pragma unroll
    for (int rr = 0; rr < 2; rr++) {
        #pragma unroll
        for (int t = 0; t < 4; t++) {
            const int col = col0 + t * 16 + l16;
            const float bb = bias[col];
            #pragma unroll
            for (int r = 0; r < 8; r++) {
                const int row = row0 + rr * 16 + half * 8 + r;
                const float v = acc[rr * 4 + t][r] + bb;
                if (MODE == 1) {
                    if (row < M) outB[(size_t)row * N + col] = (__bf16)gelu_exact(v);
                } else if (MODE == 2) {
                    const int b = row >> 11, n = row & (NPTS - 1);
                    const size_t o = ((size_t)(b * (NPTS + 1) + 1 + n)) * C_DIM + col;
                    outF[o] = resid[o] + v;
                } else { // MODE 4
                    if (row < M) {
                        const size_t o = (size_t)row * N + col;
                        outF[o] = resid[o] + v;
                    }
                }
            }
        }
    }
}

extern "C" void kernel_launch(void* const* d_in, const int* in_sizes, int n_in,
                              void* d_out, int out_size, void* d_ws, size_t ws_size,
                              hipStream_t stream) {
    const float* center  = (const float*)d_in[0];
    const float* x       = (const float*)d_in[1];
    const float* ln1_g   = (const float*)d_in[2];
    const float* ln1_b   = (const float*)d_in[3];
    const float* alpha   = (const float*)d_in[4];
    const float* beta    = (const float*)d_in[5];
    const float* attn_w1 = (const float*)d_in[6];
    const float* attn_b1 = (const float*)d_in[7];
    const float* attn_w2 = (const float*)d_in[8];
    const float* attn_b2 = (const float*)d_in[9];
    const float* ln2_g   = (const float*)d_in[10];
    const float* ln2_b   = (const float*)d_in[11];
    const float* mlp_w1  = (const float*)d_in[12];
    const float* mlp_b1  = (const float*)d_in[13];
    const float* mlp_w2  = (const float*)d_in[14];
    const float* mlp_b2  = (const float*)d_in[15];
    float* out = (float*)d_out;

    char* wp = (char*)d_ws;
    auto alloc = [&](size_t bytes) -> void* {
        void* p = (void*)wp;
        wp += (bytes + 255) & ~(size_t)255;
        return p;
    };
    float*  xn   = (float*) alloc((size_t)ROWS_ALL * C_DIM * 4);
    float*  x2   = (float*) alloc((size_t)ROWS_ALL * C_DIM * 4);
    float*  S    = (float*) alloc((size_t)ROWS_NO  * C_DIM * 4);
    int*    idx  = (int*)   alloc((size_t)ROWS_NO * KG * 4);
    float*  part = (float*) alloc((size_t)ROWS_NO * 2 * 4);
    float*  scal = (float*) alloc(256);
    __bf16* ef   = (__bf16*)alloc((size_t)ROWS_NO * 2 * C_DIM * 2);
    __bf16* H1   = (__bf16*)alloc((size_t)ROWS_NO * C_DIM * 2);
    __bf16* y    = (__bf16*)alloc((size_t)ROWS_ALL * C_DIM * 2);
    __bf16* G    = (__bf16*)alloc((size_t)ROWS_ALL * 4 * C_DIM * 2);
    __bf16* w1T  = (__bf16*)alloc((size_t)2 * C_DIM * C_DIM * 2);      // 384 x 768
    __bf16* w2T  = (__bf16*)alloc((size_t)C_DIM * C_DIM * 2);          // 384 x 384
    __bf16* mw1T = (__bf16*)alloc((size_t)4 * C_DIM * C_DIM * 2);      // 1536 x 384
    __bf16* mw2T = (__bf16*)alloc((size_t)4 * C_DIM * C_DIM * 2);      // 384 x 1536
    (void)ws_size; (void)in_sizes; (void)n_in; (void)out_size;

    // 1. LN1 -> xn (f32)
    ln_kernel<false><<<ROWS_ALL, 128, 0, stream>>>(x, ln1_g, ln1_b, xn, nullptr);
    // 2. kNN indices
    knn_kernel<<<dim3(NPTS / 256, BATCH), 256, 0, stream>>>(center, idx);
    // 3. gather + partial reductions
    gather_kernel<<<ROWS_NO, C_DIM, 0, stream>>>(xn, idx, S, part);
    // 4. global std -> inv scalar
    std_kernel<<<1, 256, 0, stream>>>(part, scal);
    // 5. ef (bf16)
    ef_kernel<<<(ROWS_NO * C_DIM + 255) / 256, 256, 0, stream>>>(S, xn, alpha, beta, scal, ef);
    // 6. weights -> bf16, N-major
    wconv_kernel<<<(2 * C_DIM * C_DIM + 255) / 256, 256, 0, stream>>>(attn_w1, w1T, 2 * C_DIM, C_DIM);
    wconv_kernel<<<(C_DIM * C_DIM + 255) / 256, 256, 0, stream>>>(attn_w2, w2T, C_DIM, C_DIM);
    wconv_kernel<<<(4 * C_DIM * C_DIM + 255) / 256, 256, 0, stream>>>(mlp_w1, mw1T, C_DIM, 4 * C_DIM);
    wconv_kernel<<<(4 * C_DIM * C_DIM + 255) / 256, 256, 0, stream>>>(mlp_w2, mw2T, 4 * C_DIM, C_DIM);
    // 7. H1 = gelu(ef @ w1 + b1)          (8192 x 768) @ (768 x 384)
    gemm_kernel<1><<<dim3(C_DIM / 64, ROWS_NO / 128), 128, 0, stream>>>(
        ef, w1T, attn_b1, ROWS_NO, 2 * C_DIM, H1, nullptr, nullptr);
    // 8. x2[.,1+n,.] = x + H1 @ w2 + b2   (8192 x 384) @ (384 x 384)
    gemm_kernel<2><<<dim3(C_DIM / 64, ROWS_NO / 128), 128, 0, stream>>>(
        H1, w2T, attn_b2, ROWS_NO, C_DIM, nullptr, x2, x);
    // 9. cls rows: x2[b,0,:] = x + LN1(x)
    cls_kernel<<<(BATCH * C_DIM + 255) / 256, 256, 0, stream>>>(x, xn, x2);
    // 10. LN2 -> y (bf16)
    ln_kernel<true><<<ROWS_ALL, 128, 0, stream>>>(x2, ln2_g, ln2_b, nullptr, y);
    // 11. G = gelu(y @ mlp_w1 + b1)       (8196 x 384) @ (384 x 1536)
    gemm_kernel<1><<<dim3(4 * C_DIM / 64, (ROWS_ALL + 127) / 128), 128, 0, stream>>>(
        y, mw1T, mlp_b1, ROWS_ALL, C_DIM, G, nullptr, nullptr);
    // 12. out = x2 + G @ mlp_w2 + b2      (8196 x 1536) @ (1536 x 384)
    gemm_kernel<4><<<dim3(C_DIM / 64, (ROWS_ALL + 127) / 128), 128, 0, stream>>>(
        G, mw2T, mlp_b2, ROWS_ALL, 4 * C_DIM, nullptr, out, x2);
}
